// MoEDecoder_86449101734128
// MI455X (gfx1250) — matrix operand order repaired and checked by reading, not compiled, not run on hardware
//
#include <hip/hip_runtime.h>
#include <hip/hip_bf16.h>

typedef __attribute__((ext_vector_type(16))) _Float16 v16h;
typedef __attribute__((ext_vector_type(8)))  _Float16 v8h;
typedef __attribute__((ext_vector_type(8)))  float    v8f;

#define B_TOT 65536
#define D_IN  128
#define H_DIM 256
#define O_DIM 128
#define E_NUM 8
#define BT    64          // batch rows per block
#define LN_EPS 1e-5f

// mish(z) = z * tanh(softplus(z)) = z * t(t+2)/(t(t+2)+2),  t = e^z
// clamp at 30: ratio == 1.0f exactly in f32; t->0 gives 0 (correct limit).
__device__ __forceinline__ float mishf(float z) {
    float t = __expf(fminf(z, 30.0f));
    float n = t * (t + 2.0f);
    return z * __fdividef(n, n + 2.0f);
}

// A fragment (16x32 f16, M = lane&15): halves 0..7  -> K = k0 + hi*16 + i
//                                      halves 8..15 -> K = k0 + hi*16 + 8 + i
__device__ __forceinline__ v16h load_a(const _Float16* row_base, int hi, int k0) {
    union { v16h v; v8h h[2]; } u;
    u.h[0] = *(const v8h*)(row_base + k0 + hi * 16);
    u.h[1] = *(const v8h*)(row_base + k0 + hi * 16 + 8);
    return u.v;
}

// ---------- preprocessing ----------
__global__ void cvt_f32_to_f16(const float* __restrict__ src,
                               _Float16* __restrict__ dst, int n) {
    int i = blockIdx.x * blockDim.x + threadIdx.x;
    int stride = gridDim.x * blockDim.x;
    for (; i < n; i += stride) dst[i] = (_Float16)src[i];
}

// src: [E][R][C] row-major  ->  dst: [E][C][R]  (transposed, f16)
__global__ void cvt_transpose(const float* __restrict__ src,
                              _Float16* __restrict__ dst, int R, int C) {
    int e = blockIdx.y;
    int n = R * C;
    const float* s = src + (size_t)e * n;
    _Float16* d = dst + (size_t)e * n;
    int i = blockIdx.x * blockDim.x + threadIdx.x;
    int stride = gridDim.x * blockDim.x;
    for (; i < n; i += stride) {
        int r = i / C, c = i - r * C;
        d[c * R + r] = (_Float16)s[i];
    }
}

// Wg [128][8] -> Wgt [16][128] f16 (transposed, zero-padded to 16 cols)
__global__ void cvt_wg(const float* __restrict__ Wg, _Float16* __restrict__ Wgt) {
    int i = blockIdx.x * blockDim.x + threadIdx.x;
    if (i < 16 * D_IN) {
        int n = i >> 7, k = i & (D_IN - 1);
        Wgt[i] = (n < E_NUM) ? (_Float16)Wg[k * E_NUM + n] : (_Float16)0.0f;
    }
}

// ---------- GEMM helper: 2 N-tiles x 4 M-tiles, fully unrolled over K ----------
__device__ __forceinline__ void gemm2x4(const _Float16* __restrict__ A, int ldk, int K,
                                        const _Float16* __restrict__ Bt, int nt0,
                                        int ln, int hi, v8f c[2][4]) {
    const _Float16* b0p = Bt + (size_t)(nt0 * 16 + ln) * ldk + hi * 16;
    const _Float16* b1p = Bt + (size_t)((nt0 + 1) * 16 + ln) * ldk + hi * 16;
    #pragma unroll
    for (int k0 = 0; k0 < K; k0 += 32) {
        __builtin_prefetch(b0p + k0 + 32, 0, 1);
        __builtin_prefetch(b1p + k0 + 32, 0, 1);
        v16h b0 = *(const v16h*)(b0p + k0);
        v16h b1 = *(const v16h*)(b1p + k0);
        v16h a[4];
        #pragma unroll
        for (int mt = 0; mt < 4; ++mt)
            a[mt] = load_a(A + (mt * 16 + ln) * ldk, hi, k0);
        #pragma unroll
        for (int mt = 0; mt < 4; ++mt)
            c[0][mt] = __builtin_amdgcn_wmma_f32_16x16x32_f16(
                           false, a[mt], false, b0, (short)0, c[0][mt], false, false);
        #pragma unroll
        for (int mt = 0; mt < 4; ++mt)
            c[1][mt] = __builtin_amdgcn_wmma_f32_16x16x32_f16(
                           false, a[mt], false, b1, (short)0, c[1][mt], false, false);
    }
}

// mish epilogue: c[2][4] (+bias) -> f16 store into dst[64][256]
__device__ __forceinline__ void mish_store(v8f c[2][4], const float* __restrict__ bias,
                                           int nt0, _Float16* __restrict__ dst,
                                           int ln, int hi) {
    #pragma unroll
    for (int t = 0; t < 2; ++t) {
        int col = (nt0 + t) * 16 + ln;
        float bb = bias[col];
        #pragma unroll
        for (int mt = 0; mt < 4; ++mt)
            #pragma unroll
            for (int r = 0; r < 8; ++r)
                dst[(mt * 16 + r + 8 * hi) * H_DIM + col] =
                    (_Float16)mishf(c[t][mt][r] + bb);
    }
}

// in-place LayerNorm over last dim (256) of sT[64][256] (f16)
__device__ __forceinline__ void ln_inplace(_Float16* sT, float* sRed,
                                           float* sMean, float* sRstd, int tid) {
    int row = tid >> 2, seg = tid & 3;
    _Float16* p = sT + row * H_DIM + seg * 64;
    float s = 0.f, s2 = 0.f;
    #pragma unroll
    for (int i = 0; i < 8; ++i) {
        v8h v = *(const v8h*)(p + i * 8);
        #pragma unroll
        for (int j = 0; j < 8; ++j) { float f = (float)v[j]; s += f; s2 += f * f; }
    }
    sRed[(0 * BT + row) * 4 + seg] = s;
    sRed[(1 * BT + row) * 4 + seg] = s2;
    __syncthreads();
    if (tid < BT) {
        float ss = 0.f, ss2 = 0.f;
        #pragma unroll
        for (int i = 0; i < 4; ++i) {
            ss  += sRed[tid * 4 + i];
            ss2 += sRed[(BT + tid) * 4 + i];
        }
        float mean = ss * (1.0f / H_DIM);
        float var  = ss2 * (1.0f / H_DIM) - mean * mean;
        sMean[tid] = mean;
        sRstd[tid] = rsqrtf(var + LN_EPS);
    }
    __syncthreads();
    float mean = sMean[row], rstd = sRstd[row];
    #pragma unroll
    for (int i = 0; i < 8; ++i) {
        v8h v = *(v8h*)(p + i * 8);
        #pragma unroll
        for (int j = 0; j < 8; ++j)
            v[j] = (_Float16)(((float)v[j] - mean) * rstd);
        *(v8h*)(p + i * 8) = v;
    }
    __syncthreads();
}

// ---------- fused MoE kernel: 64 batch rows per block, 8 waves ----------
__global__ __launch_bounds__(256) void moe_fused(
    const float* __restrict__ bg,
    const float* __restrict__ b1, const float* __restrict__ b2,
    const float* __restrict__ b3, const float* __restrict__ b4,
    const _Float16* __restrict__ xh,  const _Float16* __restrict__ Wgt,
    const _Float16* __restrict__ W1t, const _Float16* __restrict__ W2t,
    const _Float16* __restrict__ W3t, const _Float16* __restrict__ W4t,
    float* __restrict__ out)
{
    extern __shared__ char smem[];
    _Float16* sHA  = (_Float16*)(smem);                 // 64*256 f16 = 32768 B
    _Float16* sHB  = (_Float16*)(smem + 32768);         // 64*256 f16 = 32768 B
    _Float16* sX   = (_Float16*)(smem + 65536);         // 64*128 f16 = 16384 B
    float*    sGate = (float*)(smem + 81920);           // 64*8 f32   =  2048 B
    float*    sRed  = (float*)(smem + 83968);           // 2*64*4 f32 =  2048 B
    float*    sMean = (float*)(smem + 86016);           // 64 f32
    float*    sRstd = (float*)(smem + 86272);           // 64 f32     -> 86528 B total

    const int tid  = threadIdx.x;
    const int wave = tid >> 5;
    const int lane = tid & 31;
    const int ln   = lane & 15;
    const int hi   = lane >> 4;
    const int row0 = blockIdx.x * BT;

    // load x tile as f16 (8192 halves, 32 per thread)
    #pragma unroll
    for (int i = 0; i < 4; ++i)
        *(v8h*)(sX + tid * 32 + i * 8) =
            *(const v8h*)(xh + (size_t)row0 * D_IN + tid * 32 + i * 8);
    __syncthreads();

    // ---- gate logits via WMMA on wave 0: sX[64x128] @ Wgt^T[128x16] ----
    if (wave == 0) {
        v8f cg[4] = {};
        const _Float16* bp = Wgt + (size_t)ln * D_IN + hi * 16;
        #pragma unroll
        for (int k0 = 0; k0 < D_IN; k0 += 32) {
            v16h b = *(const v16h*)(bp + k0);
            #pragma unroll
            for (int mt = 0; mt < 4; ++mt) {
                v16h a = load_a(sX + (mt * 16 + ln) * D_IN, hi, k0);
                cg[mt] = __builtin_amdgcn_wmma_f32_16x16x32_f16(
                             false, a, false, b, (short)0, cg[mt], false, false);
            }
        }
        if (ln < E_NUM) {
            float bb = bg[ln];
            #pragma unroll
            for (int mt = 0; mt < 4; ++mt)
                #pragma unroll
                for (int r = 0; r < 8; ++r)
                    sGate[(mt * 16 + r + 8 * hi) * E_NUM + ln] = cg[mt][r] + bb;
        }
    }
    __syncthreads();
    if (tid < BT) {   // softmax over 8 experts per row
        float mx = sGate[tid * 8];
        for (int e = 1; e < 8; ++e) mx = fmaxf(mx, sGate[tid * 8 + e]);
        float v[8], s = 0.f;
        for (int e = 0; e < 8; ++e) { v[e] = __expf(sGate[tid * 8 + e] - mx); s += v[e]; }
        float inv = 1.0f / s;
        for (int e = 0; e < 8; ++e) sGate[tid * 8 + e] = v[e] * inv;
    }
    __syncthreads();

    float acc_out[4][8];
    #pragma unroll
    for (int mt = 0; mt < 4; ++mt)
        #pragma unroll
        for (int r = 0; r < 8; ++r) acc_out[mt][r] = 0.f;

    const int nt0 = wave * 2;

    for (int e = 0; e < E_NUM; ++e) {
        // ===== Layer 1: sX[64x128] @ W1[e] -> mish -> sHA, LN in place =====
        {
            v8f c[2][4] = {};
            gemm2x4(sX, D_IN, D_IN, W1t + (size_t)e * H_DIM * D_IN, nt0, ln, hi, c);
            mish_store(c, b1 + e * H_DIM, nt0, sHA, ln, hi);
        }
        __syncthreads();
        ln_inplace(sHA, sRed, sMean, sRstd, tid);

        // ===== Layer 2: sHA @ W2[e] -> mish -> sHB, LN in place =====
        {
            v8f c[2][4] = {};
            gemm2x4(sHA, H_DIM, H_DIM, W2t + (size_t)e * H_DIM * H_DIM, nt0, ln, hi, c);
            mish_store(c, b2 + e * H_DIM, nt0, sHB, ln, hi);
        }
        __syncthreads();
        ln_inplace(sHB, sRed, sMean, sRstd, tid);

        // ===== Layer 3: sHB @ W3[e] -> mish -> sHA (no LN) =====
        {
            v8f c[2][4] = {};
            gemm2x4(sHB, H_DIM, H_DIM, W3t + (size_t)e * H_DIM * H_DIM, nt0, ln, hi, c);
            mish_store(c, b3 + e * H_DIM, nt0, sHA, ln, hi);
        }
        __syncthreads();

        // ===== Layer 4: sHA @ W4[e] (+b4), gate-weighted accumulate =====
        {
            const _Float16* bp = W4t + (size_t)e * O_DIM * H_DIM
                               + (size_t)(wave * 16 + ln) * H_DIM + hi * 16;
            v8f c4[4] = {};
            #pragma unroll
            for (int k0 = 0; k0 < H_DIM; k0 += 32) {
                __builtin_prefetch(bp + k0 + 32, 0, 1);
                v16h b = *(const v16h*)(bp + k0);
                v16h a[4];
                #pragma unroll
                for (int mt = 0; mt < 4; ++mt)
                    a[mt] = load_a(sHA + (mt * 16 + ln) * H_DIM, hi, k0);
                #pragma unroll
                for (int mt = 0; mt < 4; ++mt)
                    c4[mt] = __builtin_amdgcn_wmma_f32_16x16x32_f16(
                                 false, a[mt], false, b, (short)0, c4[mt], false, false);
            }
            int col = wave * 16 + ln;
            float bias = b4[e * O_DIM + col];
            #pragma unroll
            for (int mt = 0; mt < 4; ++mt)
                #pragma unroll
                for (int r = 0; r < 8; ++r) {
                    int m = mt * 16 + r + 8 * hi;
                    acc_out[mt][r] += sGate[m * E_NUM + e] * (c4[mt][r] + bias);
                }
        }
        __syncthreads();   // protect sHA/sHB before next expert overwrites
    }

    // store output tile [64 x 128]
    {
        int col = wave * 16 + ln;
        #pragma unroll
        for (int mt = 0; mt < 4; ++mt)
            #pragma unroll
            for (int r = 0; r < 8; ++r) {
                int m = mt * 16 + r + 8 * hi;
                out[(size_t)(row0 + m) * O_DIM + col] = acc_out[mt][r];
            }
    }
}

extern "C" void kernel_launch(void* const* d_in, const int* in_sizes, int n_in,
                              void* d_out, int out_size, void* d_ws, size_t ws_size,
                              hipStream_t stream) {
    const float* x  = (const float*)d_in[0];
    const float* Wg = (const float*)d_in[1];
    const float* bg = (const float*)d_in[2];
    const float* W1 = (const float*)d_in[3];
    const float* b1 = (const float*)d_in[4];
    const float* W2 = (const float*)d_in[5];
    const float* b2 = (const float*)d_in[6];
    const float* W3 = (const float*)d_in[7];
    const float* b3 = (const float*)d_in[8];
    const float* W4 = (const float*)d_in[9];
    const float* b4 = (const float*)d_in[10];
    float* out = (float*)d_out;

    char* ws = (char*)d_ws;
    size_t off = 0;
    _Float16* xh  = (_Float16*)(ws + off); off += (size_t)B_TOT * D_IN * 2;
    _Float16* W1t = (_Float16*)(ws + off); off += (size_t)E_NUM * D_IN * H_DIM * 2;
    _Float16* W2t = (_Float16*)(ws + off); off += (size_t)E_NUM * H_DIM * H_DIM * 2;
    _Float16* W3t = (_Float16*)(ws + off); off += (size_t)E_NUM * H_DIM * H_DIM * 2;
    _Float16* W4t = (_Float16*)(ws + off); off += (size_t)E_NUM * H_DIM * O_DIM * 2;
    _Float16* Wgt = (_Float16*)(ws + off); off += (size_t)16 * D_IN * 2;

    cvt_f32_to_f16<<<2048, 256, 0, stream>>>(x, xh, B_TOT * D_IN);
    cvt_transpose<<<dim3(128, E_NUM), 256, 0, stream>>>(W1, W1t, D_IN,  H_DIM);
    cvt_transpose<<<dim3(256, E_NUM), 256, 0, stream>>>(W2, W2t, H_DIM, H_DIM);
    cvt_transpose<<<dim3(256, E_NUM), 256, 0, stream>>>(W3, W3t, H_DIM, H_DIM);
    cvt_transpose<<<dim3(128, E_NUM), 256, 0, stream>>>(W4, W4t, H_DIM, O_DIM);
    cvt_wg<<<8, 256, 0, stream>>>(Wg, Wgt);

    moe_fused<<<B_TOT / BT, 256, 86528, stream>>>(bg, b1, b2, b3, b4,
                                                  xh, Wgt, W1t, W2t, W3t, W4t, out);
}
